// TransformerEncoderLayer_893353198062
// MI455X (gfx1250) — compile-verified, matrix operand order checked by reading, not run on hardware
//
#include <hip/hip_runtime.h>

typedef __attribute__((ext_vector_type(16))) _Float16 v16h;
typedef __attribute__((ext_vector_type(8)))  float    v8f;
typedef __attribute__((ext_vector_type(4)))  unsigned int v4u;
typedef __attribute__((ext_vector_type(4)))  float    v4f;
typedef __attribute__((ext_vector_type(4)))  _Float16 v4h;

static constexpr int D_MODEL = 1024;
static constexpr int NH      = 16;
static constexpr int DH      = 64;
static constexpr int DFF     = 4096;
static constexpr int LSEQ    = 1024;
static constexpr int BATCH   = 4;
static constexpr float EPSV  = 1e-5f;
static constexpr float NEGV  = -1e10f;

// ---------------- GEMM config ----------------
static constexpr int BM = 128, BN = 128, BK = 32;
static constexpr int LDSTR = BK + 8;            // 40 halves = 80B row stride
static constexpr int ABUF_HALF = BM * LDSTR;    // elements per A buffer (5120)
static constexpr int BBUF_HALF = BN * LDSTR;    // elements per B buffer
static constexpr unsigned ABUF_BYTES = ABUF_HALF * 2;  // 10240

enum { EPI_F16 = 0, EPI_SCORE = 1, EPI_VT = 2, EPI_BIAS_RES_F32 = 3, EPI_BIAS_RELU_F16 = 4 };

struct GemmP {
    const _Float16* A;   // M x K, row-major (K-fast)
    const _Float16* B;   // N x K, row-major (K-fast)  ("NT" layout)
    float*          Cf;
    _Float16*       Ch;
    const float*    bias;
    const float*    resid;
    const int*      mask;
    long long sAb, sAh, sBb, sBh, sCb, sCh;   // per-batch / per-head offsets (elements)
    int lda, ldb, ldc, M, N, K;
    float scale;
};

union FragU { v16h h; v4u q[2]; };

__device__ __forceinline__ unsigned lds_raw_addr(const void* p) {
    // generic LDS pointer: bits [31:0] are the LDS byte address (aperture in high half)
    return (unsigned)(size_t)p;
}

__device__ __forceinline__ void wait_async0() {
#if defined(__has_builtin) && __has_builtin(__builtin_amdgcn_s_wait_asynccnt)
    __builtin_amdgcn_s_wait_asynccnt(0);
#else
    asm volatile("s_wait_asynccnt 0" ::: "memory");
#endif
}

// async copy of 32 contiguous bytes (global -> LDS), per lane
__device__ __forceinline__ void async_copy32(unsigned lds_addr, const _Float16* gptr) {
    asm volatile(
        "global_load_async_to_lds_b128 %0, %1, off\n\t"
        "global_load_async_to_lds_b128 %0, %1, off offset:16"
        :: "v"(lds_addr), "v"((const void*)gptr)
        : "memory");
}

template <int EPI>
__global__ __launch_bounds__(256) void gemm_wmma(GemmP p) {
    __shared__ _Float16 sA[2 * ABUF_HALF];
    __shared__ _Float16 sB[2 * BBUF_HALF];

    const int z  = blockIdx.z;
    const int bb = z / NH;
    const int hh = z % NH;
    const _Float16* gA = p.A + (size_t)bb * p.sAb + (size_t)hh * p.sAh;
    const _Float16* gB = p.B + (size_t)bb * p.sBb + (size_t)hh * p.sBh;

    const int m0 = blockIdx.y * BM;
    const int n0 = blockIdx.x * BN;

    const int tid  = threadIdx.x;
    const int wave = tid >> 5;
    const int lane = tid & 31;
    const int wm   = (wave >> 1) * 32;   // wave row origin in tile
    const int wn   = (wave & 1) * 64;    // wave col origin in tile
    const int lhalf = lane >> 4;
    const int l15   = lane & 15;

    // LDS fragment base offsets (elements, buffer 0)
    int aoff[2], boff[4];
#pragma unroll
    for (int mi = 0; mi < 2; ++mi)
        aoff[mi] = (wm + mi * 16 + l15) * LDSTR + lhalf * 8;
#pragma unroll
    for (int ni = 0; ni < 4; ++ni)
        boff[ni] = (wn + ni * 16 + l15) * LDSTR + lhalf * 8;

    v8f acc[2][4];
#pragma unroll
    for (int mi = 0; mi < 2; ++mi)
#pragma unroll
        for (int ni = 0; ni < 4; ++ni)
            acc[mi][ni] = v8f{0.f, 0.f, 0.f, 0.f, 0.f, 0.f, 0.f, 0.f};

    // tile loader indices: 256 threads, each moves 32B of A and 32B of B per tile
    const int lrow = tid >> 1;
    const int lcol = (tid & 1) << 4;   // 0 or 16 (halves)

    const bool vA = (m0 + lrow) < p.M;
    const bool vB = (n0 + lrow) < p.N;
    const _Float16* gArow = gA + (size_t)(m0 + lrow) * p.lda + lcol;
    const _Float16* gBrow = gB + (size_t)(n0 + lrow) * p.ldb + lcol;
    const unsigned lAbase = lds_raw_addr(&sA[lrow * LDSTR + lcol]);
    const unsigned lBbase = lds_raw_addr(&sB[lrow * LDSTR + lcol]);

    // zero LDS rows that the async loads will never write (invalid rows), both buffers
    {
        v4u z4 = {0u, 0u, 0u, 0u};
        if (!vA) {
            *(v4u*)&sA[lrow * LDSTR + lcol]                 = z4;
            *(v4u*)&sA[lrow * LDSTR + lcol + 8]             = z4;
            *(v4u*)&sA[ABUF_HALF + lrow * LDSTR + lcol]     = z4;
            *(v4u*)&sA[ABUF_HALF + lrow * LDSTR + lcol + 8] = z4;
        }
        if (!vB) {
            *(v4u*)&sB[lrow * LDSTR + lcol]                 = z4;
            *(v4u*)&sB[lrow * LDSTR + lcol + 8]             = z4;
            *(v4u*)&sB[BBUF_HALF + lrow * LDSTR + lcol]     = z4;
            *(v4u*)&sB[BBUF_HALF + lrow * LDSTR + lcol + 8] = z4;
        }
    }

    auto issue_tile = [&](int k0, int buf) {
        if (vA) async_copy32(lAbase + buf * ABUF_BYTES, gArow + k0);
        if (vB) async_copy32(lBbase + buf * ABUF_BYTES, gBrow + k0);
    };

    const int nk = p.K / BK;
    issue_tile(0, 0);

    int buf = 0;
    for (int i = 0; i < nk; ++i) {
        wait_async0();        // our tile-i async loads have landed in LDS
        __syncthreads();      // everyone's landed; everyone is done reading buf^1
        if (i + 1 < nk) issue_tile((i + 1) * BK, buf ^ 1);

        const int ab = buf * ABUF_HALF;
        const int bbs = buf * BBUF_HALF;
        FragU fa[2], fb[4];
#pragma unroll
        for (int mi = 0; mi < 2; ++mi) {
            fa[mi].q[0] = *(const v4u*)&sA[ab + aoff[mi]];
            fa[mi].q[1] = *(const v4u*)&sA[ab + aoff[mi] + 16];
        }
#pragma unroll
        for (int ni = 0; ni < 4; ++ni) {
            fb[ni].q[0] = *(const v4u*)&sB[bbs + boff[ni]];
            fb[ni].q[1] = *(const v4u*)&sB[bbs + boff[ni] + 16];
        }
#pragma unroll
        for (int mi = 0; mi < 2; ++mi)
#pragma unroll
            for (int ni = 0; ni < 4; ++ni)
                acc[mi][ni] = __builtin_amdgcn_wmma_f32_16x16x32_f16(
                    false, fa[mi].h, false, fb[ni].h, (short)0, acc[mi][ni], false, false);
        buf ^= 1;
    }

    const size_t cb = (size_t)bb * p.sCb + (size_t)hh * p.sCh;
#pragma unroll
    for (int mi = 0; mi < 2; ++mi) {
#pragma unroll
        for (int ni = 0; ni < 4; ++ni) {
#pragma unroll
            for (int r = 0; r < 8; ++r) {
                const int m = m0 + wm + mi * 16 + (lhalf << 3) + r;
                const int n = n0 + wn + ni * 16 + l15;
                if (m >= p.M || n >= p.N) continue;
                float v = acc[mi][ni][r];
                if constexpr (EPI == EPI_F16) {
                    p.Ch[cb + (size_t)m * p.ldc + n] = (_Float16)v;
                } else if constexpr (EPI == EPI_SCORE) {
                    v *= p.scale;
                    if (p.mask[bb * LSEQ + n] == 0) v = NEGV;
                    p.Ch[cb + (size_t)m * p.ldc + n] = (_Float16)v;
                } else if constexpr (EPI == EPI_VT) {
                    // V^T per head: out[((b*1024 + n) << 10) | l], b = m/1024, l = m%1024
                    const size_t idx = ((size_t)((m >> 10) * 1024 + n) << 10) | (size_t)(m & 1023);
                    p.Ch[idx] = (_Float16)v;
                } else if constexpr (EPI == EPI_BIAS_RES_F32) {
                    v += p.bias[n] + p.resid[(size_t)m * p.ldc + n];
                    p.Cf[(size_t)m * p.ldc + n] = v;
                } else if constexpr (EPI == EPI_BIAS_RELU_F16) {
                    v += p.bias[n];
                    v = v > 0.f ? v : 0.f;
                    p.Ch[(size_t)m * p.ldc + n] = (_Float16)v;
                }
            }
        }
    }
}

// ---------------- helper kernels ----------------

__global__ __launch_bounds__(256) void cvt_f32_to_f16(const float* __restrict__ x,
                                                      _Float16* __restrict__ y, int n) {
    int i = (blockIdx.x * 256 + threadIdx.x) * 4;
    if (i + 3 < n) {
        v4f f = *(const v4f*)(x + i);
        *(v4h*)(y + i) = __builtin_convertvector(f, v4h);
    }
}

// W (K x N, row-major f32) -> Wt (N x K, row-major f16)
__global__ __launch_bounds__(256) void transpose_cvt(const float* __restrict__ W,
                                                     _Float16* __restrict__ Wt, int K, int N) {
    __shared__ float t[32][33];
    const int n  = blockIdx.x * 32 + threadIdx.x;
    const int k0 = blockIdx.y * 32;
    for (int i = threadIdx.y; i < 32; i += 8)
        t[i][threadIdx.x] = W[(size_t)(k0 + i) * N + n];
    __syncthreads();
    const int k  = k0 + threadIdx.x;
    const int nb = blockIdx.x * 32;
    for (int i = threadIdx.y; i < 32; i += 8)
        Wt[(size_t)(nb + i) * K + k] = (_Float16)t[threadIdx.x][i];
}

// in-place row softmax over f16 rows of length 1024 (mask already baked in)
__global__ __launch_bounds__(256) void softmax_rows(_Float16* __restrict__ S) {
    __shared__ float red[256];
    const int t = threadIdx.x;
    _Float16* row = S + (size_t)blockIdx.x * LSEQ;
    v4h hv = *(const v4h*)(row + t * 4);
    float v[4];
#pragma unroll
    for (int i = 0; i < 4; ++i) v[i] = (float)hv[i];

    float mx = fmaxf(fmaxf(v[0], v[1]), fmaxf(v[2], v[3]));
    red[t] = mx;
    __syncthreads();
    for (int s = 128; s > 0; s >>= 1) {
        if (t < s) red[t] = fmaxf(red[t], red[t + s]);
        __syncthreads();
    }
    const float rmax = red[0];
    __syncthreads();

    float e[4], acc = 0.f;
#pragma unroll
    for (int i = 0; i < 4; ++i) { e[i] = __expf(v[i] - rmax); acc += e[i]; }
    red[t] = acc;
    __syncthreads();
    for (int s = 128; s > 0; s >>= 1) {
        if (t < s) red[t] += red[t + s];
        __syncthreads();
    }
    const float inv = 1.f / red[0];

    v4h o;
#pragma unroll
    for (int i = 0; i < 4; ++i) o[i] = (_Float16)(e[i] * inv);
    *(v4h*)(row + t * 4) = o;
}

// per-row LayerNorm over 1024 f32; write f32 out and optionally f16 copy
template <bool WRITE_H>
__global__ __launch_bounds__(256) void layernorm_row(const float* __restrict__ X,
                                                     const float* __restrict__ g,
                                                     const float* __restrict__ be,
                                                     float* __restrict__ Y,
                                                     _Float16* __restrict__ Yh) {
    __shared__ float red[256];
    const int t = threadIdx.x;
    const float* row = X + (size_t)blockIdx.x * D_MODEL;
    v4f v = *(const v4f*)(row + t * 4);

    float s = v[0] + v[1] + v[2] + v[3];
    red[t] = s;
    __syncthreads();
    for (int st = 128; st > 0; st >>= 1) {
        if (t < st) red[t] += red[t + st];
        __syncthreads();
    }
    const float mu = red[0] * (1.f / D_MODEL);
    __syncthreads();

    float sq = 0.f;
#pragma unroll
    for (int i = 0; i < 4; ++i) { float d = v[i] - mu; sq += d * d; }
    red[t] = sq;
    __syncthreads();
    for (int st = 128; st > 0; st >>= 1) {
        if (t < st) red[t] += red[t + st];
        __syncthreads();
    }
    const float rs = rsqrtf(red[0] * (1.f / D_MODEL) + EPSV);

    v4f gg = *(const v4f*)(g + t * 4);
    v4f bb = *(const v4f*)(be + t * 4);
    v4f o;
#pragma unroll
    for (int i = 0; i < 4; ++i) o[i] = (v[i] - mu) * rs * gg[i] + bb[i];
    *(v4f*)(Y + (size_t)blockIdx.x * D_MODEL + t * 4) = o;
    if constexpr (WRITE_H) {
        *(v4h*)(Yh + (size_t)blockIdx.x * D_MODEL + t * 4) = __builtin_convertvector(o, v4h);
    }
}

// ---------------- host orchestration ----------------

extern "C" void kernel_launch(void* const* d_in, const int* in_sizes, int n_in,
                              void* d_out, int out_size, void* d_ws, size_t ws_size,
                              hipStream_t stream) {
    (void)in_sizes; (void)n_in; (void)out_size; (void)ws_size;

    const float* x    = (const float*)d_in[0];
    const int*   mask = (const int*)d_in[1];
    const float* Wq   = (const float*)d_in[2];
    const float* Wk   = (const float*)d_in[3];
    const float* Wv   = (const float*)d_in[4];
    const float* Wp   = (const float*)d_in[5];
    const float* bp   = (const float*)d_in[6];
    const float* W1   = (const float*)d_in[7];
    const float* b1   = (const float*)d_in[8];
    const float* W2   = (const float*)d_in[9];
    const float* b2   = (const float*)d_in[10];
    const float* g1   = (const float*)d_in[11];
    const float* be1  = (const float*)d_in[12];
    const float* g2   = (const float*)d_in[13];
    const float* be2  = (const float*)d_in[14];
    float* out = (float*)d_out;

    const size_t NTOK = (size_t)BATCH * LSEQ;            // 4096 tokens
    const size_t NX   = NTOK * D_MODEL;                  // 4Mi activations

    char* ws = (char*)d_ws;
    size_t off = 0;
    auto alloc = [&](size_t bytes) -> void* {
        void* p = ws + off;
        off += (bytes + 255) & ~(size_t)255;
        return p;
    };

    _Float16* xb  = (_Float16*)alloc(NX * 2);
    _Float16* Wqt = (_Float16*)alloc((size_t)D_MODEL * D_MODEL * 2);
    _Float16* Wkt = (_Float16*)alloc((size_t)D_MODEL * D_MODEL * 2);
    _Float16* Wvt = (_Float16*)alloc((size_t)D_MODEL * D_MODEL * 2);
    _Float16* Wpt = (_Float16*)alloc((size_t)D_MODEL * D_MODEL * 2);
    _Float16* W1t = (_Float16*)alloc((size_t)D_MODEL * DFF * 2);
    _Float16* W2t = (_Float16*)alloc((size_t)DFF * D_MODEL * 2);
    _Float16* Qb  = (_Float16*)alloc(NX * 2);
    _Float16* Kb  = (_Float16*)alloc(NX * 2);
    _Float16* Vt  = (_Float16*)alloc(NX * 2);
    _Float16* S   = (_Float16*)alloc((size_t)BATCH * NH * LSEQ * LSEQ * 2);
    _Float16* Hd  = (_Float16*)alloc(NX * 2);
    float*    AO  = (float*)alloc(NX * 4);
    float*    Hf  = (float*)alloc(NX * 4);
    _Float16* Hh  = (_Float16*)alloc(NX * 2);
    _Float16* F1  = (_Float16*)alloc(NTOK * (size_t)DFF * 2);
    float*    T2  = (float*)alloc(NX * 4);

    // 1) x -> f16
    cvt_f32_to_f16<<<dim3((unsigned)(NX / 1024)), 256, 0, stream>>>(x, xb, (int)NX);

    // 2) weights -> transposed f16 (N x K)
    dim3 tb(32, 8);
    transpose_cvt<<<dim3(32, 32), tb, 0, stream>>>(Wq, Wqt, D_MODEL, D_MODEL);
    transpose_cvt<<<dim3(32, 32), tb, 0, stream>>>(Wk, Wkt, D_MODEL, D_MODEL);
    transpose_cvt<<<dim3(32, 32), tb, 0, stream>>>(Wv, Wvt, D_MODEL, D_MODEL);
    transpose_cvt<<<dim3(32, 32), tb, 0, stream>>>(Wp, Wpt, D_MODEL, D_MODEL);
    transpose_cvt<<<dim3(128, 32), tb, 0, stream>>>(W1, W1t, D_MODEL, DFF);
    transpose_cvt<<<dim3(32, 128), tb, 0, stream>>>(W2, W2t, DFF, D_MODEL);

    auto mkP = [](const _Float16* A, int lda, const _Float16* B, int ldb,
                  float* Cf, _Float16* Ch, int ldc, int M, int N, int K) {
        GemmP p{};
        p.A = A; p.B = B; p.Cf = Cf; p.Ch = Ch;
        p.bias = nullptr; p.resid = nullptr; p.mask = nullptr;
        p.sAb = p.sAh = p.sBb = p.sBh = p.sCb = p.sCh = 0;
        p.lda = lda; p.ldb = ldb; p.ldc = ldc; p.M = M; p.N = N; p.K = K;
        p.scale = 1.f;
        return p;
    };

    // 3) Q = xb @ Wq^T-layout, K = ..., V -> transposed per head
    {
        GemmP p = mkP(xb, D_MODEL, Wqt, D_MODEL, nullptr, Qb, D_MODEL,
                      (int)NTOK, D_MODEL, D_MODEL);
        gemm_wmma<EPI_F16><<<dim3(8, 32, 1), 256, 0, stream>>>(p);
        p.B = Wkt; p.Ch = Kb;
        gemm_wmma<EPI_F16><<<dim3(8, 32, 1), 256, 0, stream>>>(p);
        p.B = Wvt; p.Ch = Vt;
        gemm_wmma<EPI_VT><<<dim3(8, 32, 1), 256, 0, stream>>>(p);
    }

    // 4) scores S[b,h] = (Q_h @ K_h^T) / 8, masked, stored f16
    {
        GemmP p = mkP(Qb, D_MODEL, Kb, D_MODEL, nullptr, S, LSEQ, LSEQ, LSEQ, DH);
        p.sAb = (long long)LSEQ * D_MODEL; p.sAh = DH;
        p.sBb = (long long)LSEQ * D_MODEL; p.sBh = DH;
        p.sCb = (long long)NH * LSEQ * LSEQ; p.sCh = (long long)LSEQ * LSEQ;
        p.mask = mask; p.scale = 0.125f;
        gemm_wmma<EPI_SCORE><<<dim3(8, 8, BATCH * NH), 256, 0, stream>>>(p);
    }

    // 5) softmax rows (in place)
    softmax_rows<<<dim3(BATCH * NH * LSEQ), 256, 0, stream>>>(S);

    // 6) heads: Hd[b,l,h*64+d] = P[b,h] @ V[b,h]
    {
        GemmP p = mkP(S, LSEQ, Vt, LSEQ, nullptr, Hd, D_MODEL, LSEQ, DH, LSEQ);
        p.sAb = (long long)NH * LSEQ * LSEQ; p.sAh = (long long)LSEQ * LSEQ;
        p.sBb = (long long)NH * DH * LSEQ;   p.sBh = (long long)DH * LSEQ;
        p.sCb = (long long)LSEQ * D_MODEL;   p.sCh = DH;
        gemm_wmma<EPI_F16><<<dim3(1, 8, BATCH * NH), 256, 0, stream>>>(p);
    }

    // 7) attn_out = Hd @ Wp + bp + x   (f32)
    {
        GemmP p = mkP(Hd, D_MODEL, Wpt, D_MODEL, AO, nullptr, D_MODEL,
                      (int)NTOK, D_MODEL, D_MODEL);
        p.bias = bp; p.resid = x;
        gemm_wmma<EPI_BIAS_RES_F32><<<dim3(8, 32, 1), 256, 0, stream>>>(p);
    }

    // 8) h = LN(attn_out) -> Hf (f32) + Hh (f16)
    layernorm_row<true><<<dim3((unsigned)NTOK), 256, 0, stream>>>(AO, g1, be1, Hf, Hh);

    // 9) F1 = relu(h @ W1 + b1)  (f16)
    {
        GemmP p = mkP(Hh, D_MODEL, W1t, D_MODEL, nullptr, F1, DFF,
                      (int)NTOK, DFF, D_MODEL);
        p.bias = b1;
        gemm_wmma<EPI_BIAS_RELU_F16><<<dim3(32, 32, 1), 256, 0, stream>>>(p);
    }

    // 10) T2 = F1 @ W2 + b2 + h  (f32)
    {
        GemmP p = mkP(F1, DFF, W2t, DFF, T2, nullptr, D_MODEL,
                      (int)NTOK, D_MODEL, DFF);
        p.bias = b2; p.resid = Hf;
        gemm_wmma<EPI_BIAS_RES_F32><<<dim3(8, 32, 1), 256, 0, stream>>>(p);
    }

    // 11) out = LN(T2)
    layernorm_row<false><<<dim3((unsigned)NTOK), 256, 0, stream>>>(T2, g2, be2, out, nullptr);
}
